// PointSAModuleMSGMask_5892695130406
// MI455X (gfx1250) — matrix operand order repaired and checked by reading, not compiled, not run on hardware
//
#include <hip/hip_runtime.h>

// ---------------------------------------------------------------------------
// PointNet++ SA-MSG module for MI455X (gfx1250, wave32, WMMA).
// Pipeline:
//   [weights f32->bf16, K-padded, layer0 col-permuted]
//   [features [B,C,N] f32 -> [B,N,C] bf16 transpose (L2-resident rows)]
//   [gather new_xyz] -> [wave-ballot 3-radius ball query] ->
//   [per-scale: async Global->LDS gather of 128 columns + 3x WMMA bf16 MLP
//    layers (A fragments register-cached across all 8 column tiles)
//    + ReLU + max-over-S via LDS ds_max]
// ---------------------------------------------------------------------------

typedef __attribute__((ext_vector_type(2)))  __bf16 bf16x2;
typedef __attribute__((ext_vector_type(8)))  __bf16 bf16x8;
typedef __attribute__((ext_vector_type(16))) __bf16 v16bf;
typedef __attribute__((ext_vector_type(8)))  float  v8f;

static constexpr int Bn    = 4;
static constexpr int Np    = 16384;
static constexpr int Cf    = 128;
static constexpr int Mq    = 1024;
static constexpr int FEATC = 640;        // 128 + 256 + 256
static constexpr int KP0   = 160;        // 131 padded to multiple of 32
static constexpr int COLS  = 128;        // columns (m,s pairs) per block

__device__ __forceinline__ __bf16 f2bf(float f) {
    unsigned u = __float_as_uint(f);
    unsigned r = (u + 0x7FFFu + ((u >> 16) & 1u)) >> 16;   // RNE
    unsigned short hs = (unsigned short)r;
    return __builtin_bit_cast(__bf16, hs);
}

__device__ __forceinline__ unsigned pack2bf(float lo, float hi) {
#if __has_builtin(__builtin_amdgcn_cvt_pk_bf16_f32)
    bf16x2 p = __builtin_amdgcn_cvt_pk_bf16_f32(lo, hi);
    return __builtin_bit_cast(unsigned, p);
#else
    unsigned a = __builtin_bit_cast(unsigned short, f2bf(lo));
    unsigned b = __builtin_bit_cast(unsigned short, f2bf(hi));
    return a | (b << 16);
#endif
}

// ---------------------------------------------------------------------------
// Kernel 0: weights -> bf16, K padded with zeros. Layer-0 matrices get their
// input columns permuted to the new X row order [feat(128), xyz(3), pad(29)].
// ---------------------------------------------------------------------------
struct WSeg  { const float* src; int rows; int cin; int kpad; int dstOff; int perm; };
struct WSegs { WSeg s[9]; };

__global__ void convert_weights_kernel(WSegs segs, __bf16* __restrict__ dst) {
    int tid = blockIdx.x * blockDim.x + threadIdx.x;
    int stride = gridDim.x * blockDim.x;
    for (int si = 0; si < 9; ++si) {
        WSeg sg = segs.s[si];
        int tot = sg.rows * sg.kpad;
        for (int e = tid; e < tot; e += stride) {
            int r = e / sg.kpad, k = e % sg.kpad;
            int srcc;
            if (sg.perm)  srcc = (k < 128) ? (k + 3) : ((k < 131) ? (k - 128) : -1);
            else          srcc = (k < sg.cin) ? k : -1;
            float v = (srcc >= 0) ? sg.src[(size_t)r * sg.cin + srcc] : 0.0f;
            dst[(size_t)sg.dstOff + e] = f2bf(v);
        }
    }
}

// ---------------------------------------------------------------------------
// Kernel 1: features [B,C,N] f32 -> featT [B,N,C] bf16 via LDS tile transpose
// (coalesced on both sides). 16.8 MB total, L2-resident afterwards.
// ---------------------------------------------------------------------------
__global__ __launch_bounds__(256)
void transpose_feat_kernel(const float* __restrict__ features,
                           __bf16* __restrict__ featT) {
    __shared__ __bf16 tile[64][Cf + 2];
    int b  = blockIdx.x / (Np / 64);
    int n0 = (blockIdx.x % (Np / 64)) * 64;
    int tid = threadIdx.x;
    for (int e = tid; e < 64 * Cf; e += 256) {
        int c = e >> 6, n = e & 63;                     // coalesced over n
        tile[n][c] = f2bf(features[((size_t)b * Cf + c) * Np + n0 + n]);
    }
    __syncthreads();
    int r = tid >> 2, qd = tid & 3;                     // 64 rows x 4 chunks
    __bf16* dst = featT + ((size_t)b * Np + n0 + r) * Cf + qd * 32;
    #pragma unroll
    for (int i = 0; i < 4; ++i) {
        bf16x8 vv;
        #pragma unroll
        for (int u = 0; u < 8; ++u) vv[u] = tile[r][qd * 32 + i * 8 + u];
        *reinterpret_cast<bf16x8*>(dst + i * 8) = vv;
    }
}

// ---------------------------------------------------------------------------
// Kernel 2: new_xyz gather; also emit new_xyz and indices portions of d_out.
// ---------------------------------------------------------------------------
__global__ void gather_newxyz_kernel(const float* __restrict__ points,
                                     const int*   __restrict__ indices,
                                     float* __restrict__ nxyz_ws,
                                     float* __restrict__ out_xyz,
                                     float* __restrict__ out_idx) {
    int t = blockIdx.x * blockDim.x + threadIdx.x;
    if (t >= Bn * Mq) return;
    int b = t / Mq;
    int i = indices[t];
    size_t p = ((size_t)b * Np + i) * 3;
    float x = points[p], y = points[p + 1], z = points[p + 2];
    nxyz_ws[t * 3 + 0] = x; nxyz_ws[t * 3 + 1] = y; nxyz_ws[t * 3 + 2] = z;
    out_xyz[t * 3 + 0] = x; out_xyz[t * 3 + 1] = y; out_xyz[t * 3 + 2] = z;
    out_idx[t] = (float)i;
}

// ---------------------------------------------------------------------------
// Kernel 3: fused 3-radius ball query. One wave32 per query; ballot
// compaction keeps lowest-index-first semantics; pads with first hit (or 0).
// ---------------------------------------------------------------------------
__global__ __launch_bounds__(256)
void ball_query_kernel(const float* __restrict__ points,
                       const float* __restrict__ nxyz,
                       int* __restrict__ idx0, int* __restrict__ idx1,
                       int* __restrict__ idx2) {
    int wid  = blockIdx.x * (blockDim.x >> 5) + (threadIdx.x >> 5);
    int lane = threadIdx.x & 31;
    int b = wid / Mq;
    float nx = nxyz[wid * 3 + 0], ny = nxyz[wid * 3 + 1], nz = nxyz[wid * 3 + 2];
    const float* pb = points + (size_t)b * Np * 3;
    int* o0 = idx0 + (size_t)wid * 16;
    int* o1 = idx1 + (size_t)wid * 32;
    int* o2 = idx2 + (size_t)wid * 64;
    int c0 = 0, c1 = 0, c2 = 0, f0 = 0, f1 = 0, f2 = 0;
    unsigned below = (1u << lane) - 1u;

    for (int base = 0; base < Np; base += 32) {
        int n = base + lane;
        float dx = pb[n * 3 + 0] - nx;
        float dy = pb[n * 3 + 1] - ny;
        float dz = pb[n * 3 + 2] - nz;
        float d2 = dx * dx + dy * dy + dz * dz;
        unsigned mk2 = __builtin_amdgcn_ballot_w32(d2 < 4.0f);
        unsigned mk1 = __builtin_amdgcn_ballot_w32(d2 < 1.0f);
        unsigned mk0 = __builtin_amdgcn_ballot_w32(d2 < 0.25f);
        if (mk0) {
            if (c0 == 0) f0 = base + __builtin_ctz(mk0);
            int pos = c0 + __builtin_popcount(mk0 & below);
            if (((mk0 >> lane) & 1u) && pos < 16) o0[pos] = n;
            c0 += __builtin_popcount(mk0);
        }
        if (mk1) {
            if (c1 == 0) f1 = base + __builtin_ctz(mk1);
            int pos = c1 + __builtin_popcount(mk1 & below);
            if (((mk1 >> lane) & 1u) && pos < 32) o1[pos] = n;
            c1 += __builtin_popcount(mk1);
        }
        if (mk2) {
            if (c2 == 0) f2 = base + __builtin_ctz(mk2);
            int pos = c2 + __builtin_popcount(mk2 & below);
            if (((mk2 >> lane) & 1u) && pos < 64) o2[pos] = n;
            c2 += __builtin_popcount(mk2);
        }
        if (c0 >= 16 && c1 >= 32 && c2 >= 64) break;
    }
    int p0 = (c0 > 0) ? f0 : 0;
    int p1 = (c1 > 0) ? f1 : 0;
    int p2 = (c2 > 0) ? f2 : 0;
    for (int t = c0 + lane; t < 16; t += 32) o0[t] = p0;
    for (int t = c1 + lane; t < 32; t += 32) o1[t] = p1;
    for (int t = c2 + lane; t < 64; t += 32) o2[t] = p2;
}

// ---------------------------------------------------------------------------
// WMMA GEMM layer over COLS LDS columns. W: global bf16 [ROWS][KDIM].
// Each wave owns whole row-tiles: A fragments (full K) are loaded ONCE into
// registers, then reused across all COLS/16 column tiles (weights are the
// dominant L2 stream; this is the 4-8x traffic reduction).
// Epilogue: ReLU(acc*g + b); paired bf16 pack + b32 LDS store, or
// ds_max-reduce (int-bitpattern max, valid for non-negative floats).
// ---------------------------------------------------------------------------
template<int KDIM, int ROWS>
static __device__ __forceinline__ void gemm_layer(
        const __bf16* __restrict__ W,
        const __bf16* X, int xpitch,
        const float* __restrict__ gamma, const float* __restrict__ beta,
        __bf16* Y, int ypitch,
        int* outMax, int sdim, int coutDim) {
    const int lane = threadIdx.x & 31;
    const int wave = threadIdx.x >> 5;
    const int hi   = lane >> 4;
    constexpr int KT = KDIM / 32;
    constexpr int RT = ROWS / 16;
    constexpr int CT = COLS / 16;
    for (int rt = wave; rt < RT; rt += 8) {
        // ---- A fragments for this row-tile, all K, register-resident ----
        int arow = rt * 16 + (lane & 15);
        v16bf A[KT];
        #pragma unroll
        for (int kt = 0; kt < KT; ++kt) {
            int akb = kt * 32 + hi * 8;
            bf16x8 a0 = *reinterpret_cast<const bf16x8*>(W + (size_t)arow * KDIM + akb);
            bf16x8 a1 = *reinterpret_cast<const bf16x8*>(W + (size_t)arow * KDIM + akb + 16);
            A[kt] = __builtin_shufflevector(a0, a1, 0,1,2,3,4,5,6,7,8,9,10,11,12,13,14,15);
        }
        int mbase = rt * 16 + hi * 8;   // C/D layout: VGPR v -> M = mbase+v
        float4 gl0 = *reinterpret_cast<const float4*>(gamma + mbase);
        float4 gl1 = *reinterpret_cast<const float4*>(gamma + mbase + 4);
        float4 bl0 = *reinterpret_cast<const float4*>(beta + mbase);
        float4 bl1 = *reinterpret_cast<const float4*>(beta + mbase + 4);
        float gv[8] = {gl0.x, gl0.y, gl0.z, gl0.w, gl1.x, gl1.y, gl1.z, gl1.w};
        float bv[8] = {bl0.x, bl0.y, bl0.z, bl0.w, bl1.x, bl1.y, bl1.z, bl1.w};

        #pragma unroll
        for (int ct = 0; ct < CT; ++ct) {
            int bn = ct * 16 + (lane & 15);
            v8f acc = {0, 0, 0, 0, 0, 0, 0, 0};
            #pragma unroll
            for (int kt = 0; kt < KT; ++kt) {
                int bk = kt * 32 + hi * 8;
                const __bf16* xp = X + bn * xpitch + bk;
                bf16x8 b0v = *reinterpret_cast<const bf16x8*>(xp);
                bf16x8 b1v = *reinterpret_cast<const bf16x8*>(xp + 16);
                v16bf Bv = __builtin_shufflevector(b0v, b1v,
                               0,1,2,3,4,5,6,7,8,9,10,11,12,13,14,15);
                acc = __builtin_amdgcn_wmma_f32_16x16x32_bf16(
                          false, A[kt], false, Bv, (short)0, acc, false, false);
            }
            if (outMax) {
                int qq = bn / sdim;
                #pragma unroll
                for (int v = 0; v < 8; ++v) {
                    float val = fmaxf(acc[v] * gv[v] + bv[v], 0.0f);
                    atomicMax(&outMax[qq * coutDim + mbase + v], __float_as_int(val));
                }
            } else {
                #pragma unroll
                for (int v = 0; v < 8; v += 2) {
                    float lo = fmaxf(acc[v]     * gv[v]     + bv[v],     0.0f);
                    float hh = fmaxf(acc[v + 1] * gv[v + 1] + bv[v + 1], 0.0f);
                    *reinterpret_cast<unsigned*>(Y + bn * ypitch + mbase + v) =
                        pack2bf(lo, hh);
                }
            }
        }
    }
}

// ---------------------------------------------------------------------------
// Kernel 4 (per scale): async Global->LDS gather of 128 columns (bf16 rows of
// featT, 256B each; X rows = [feat 0..127 | xyz 128..130 | zero pad]), then
// 3 WMMA layers, max-over-S epilogue, write output.
// ---------------------------------------------------------------------------
template<int S, int H, int COUT>
__global__ __launch_bounds__(256)
void fused_group_mlp_kernel(const float* __restrict__ points,
                            const __bf16* __restrict__ featT,
                            const float* __restrict__ nxyz,
                            const int*   __restrict__ idx,
                            const __bf16* __restrict__ wbf,
                            int w0off, int w1off, int w2off,
                            const float* g0, const float* b0,
                            const float* g1, const float* b1,
                            const float* g2, const float* b2,
                            float* __restrict__ outFeat, int cOff) {
    constexpr int Q = COLS / S;           // queries per block
    __shared__ __bf16 xT [COLS][KP0];     // [col][feat 0..127 | xyz | pad]
    __shared__ __bf16 a1T[COLS][H];       // layer-1 activations (transposed)
    __shared__ __bf16 a2T[COLS][H];       // layer-2 activations (transposed)
    __shared__ int    outMax[Q][COUT];    // max-over-S accumulator (f32 bits)

    const int tid = threadIdx.x;
    const int blocksPerB = Mq / Q;
    const int b  = blockIdx.x / blocksPerB;
    const int m0 = (blockIdx.x % blocksPerB) * Q;

    for (int e = tid; e < Q * COUT; e += 256) (&outMax[0][0])[e] = 0; // == 0.0f

    // ---- gather: 128 cols x 2 lanes; 8x async b128 (128B) per lane ----
    {
        int j = tid & (COLS - 1), part = tid >> 7;      // part in {0,1}
        int q = j / S, s = j % S;
        int m = m0 + q;
        int pidx = idx[(size_t)(b * Mq + m) * S + s];
        const __bf16* src = featT + ((size_t)b * Np + pidx) * Cf + part * 64;
        // flat address of LDS object: low 32 bits are the wave-relative LDS
        // byte address (aperture mapping truncates addr[31:0]).
        unsigned ldsa = (unsigned)(size_t)&xT[j][part * 64];
        #pragma unroll
        for (int i = 0; i < 8; ++i) {
            asm volatile("global_load_async_to_lds_b128 %0, %1, off"
                         :: "v"(ldsa + i * 16), "v"(src + i * 8)
                         : "memory");
        }
        if (part == 0) {
            size_t pp = ((size_t)b * Np + pidx) * 3;
            size_t nn = (size_t)(b * Mq + m) * 3;
            xT[j][128] = f2bf(points[pp + 0] - nxyz[nn + 0]);
            xT[j][129] = f2bf(points[pp + 1] - nxyz[nn + 1]);
            xT[j][130] = f2bf(points[pp + 2] - nxyz[nn + 2]);
            for (int k = 131; k < KP0; ++k) xT[j][k] = f2bf(0.0f);
        }
        asm volatile("s_wait_asynccnt 0x0" ::: "memory");
    }
    __syncthreads();

    gemm_layer<KP0, H>(wbf + w0off, &xT[0][0],  KP0, g0, b0,
                       &a1T[0][0], H, nullptr, S, COUT);
    __syncthreads();
    gemm_layer<H, H>  (wbf + w1off, &a1T[0][0], H,   g1, b1,
                       &a2T[0][0], H, nullptr, S, COUT);
    __syncthreads();
    gemm_layer<H, COUT>(wbf + w2off, &a2T[0][0], H,  g2, b2,
                        nullptr, 0, &outMax[0][0], S, COUT);
    __syncthreads();

    for (int e = tid; e < Q * COUT; e += 256) {
        int q = e / COUT, c = e % COUT;
        float val = __int_as_float(outMax[q][c]);
        outFeat[((size_t)b * FEATC + (cOff + c)) * Mq + (m0 + q)] = val;
    }
}

// ---------------------------------------------------------------------------
// Host-side orchestration.
// ---------------------------------------------------------------------------
extern "C" void kernel_launch(void* const* d_in, const int* in_sizes, int n_in,
                              void* d_out, int out_size, void* d_ws, size_t ws_size,
                              hipStream_t stream) {
    (void)in_sizes; (void)n_in; (void)out_size; (void)ws_size;
    const float* points   = (const float*)d_in[0];
    const float* features = (const float*)d_in[1];
    const int*   indices  = (const int*)d_in[2];
    auto W = [&](int s, int l) { return (const float*)d_in[3 + (s * 3 + l) * 3 + 0]; };
    auto G = [&](int s, int l) { return (const float*)d_in[3 + (s * 3 + l) * 3 + 1]; };
    auto Bb= [&](int s, int l) { return (const float*)d_in[3 + (s * 3 + l) * 3 + 2]; };

    // workspace layout (bytes)
    char* ws = (char*)d_ws;
    __bf16* wbf   = reinterpret_cast<__bf16*>(ws);           // 161792 bf16 = 323584 B
    float*  nxyz  = reinterpret_cast<float*>(ws + 323584);   // B*M*3 f32   =  49152 B
    int*    idx0  = reinterpret_cast<int*>(ws + 372736);     // B*M*16
    int*    idx1  = idx0 + Bn * Mq * 16;                     // B*M*32
    int*    idx2  = idx1 + Bn * Mq * 32;                     // B*M*64
    __bf16* featT = reinterpret_cast<__bf16*>(ws + 2207744); // B*N*C bf16 = 16 MiB

    float* out      = (float*)d_out;
    float* out_xyz  = out;                                   // [B,M,3]
    float* out_feat = out + Bn * Mq * 3;                     // [B,640,M]
    float* out_idx  = out_feat + (size_t)Bn * FEATC * Mq;    // [B,M]

    // bf16 weight segments (rows, cin, kpad, dstOff elems, perm-layer0 flag)
    WSegs segs;
    segs.s[0] = { W(0,0),  64, 131, 160,      0, 1 };
    segs.s[1] = { W(0,1),  64,  64,  64,  10240, 0 };
    segs.s[2] = { W(0,2), 128,  64,  64,  14336, 0 };
    segs.s[3] = { W(1,0), 128, 131, 160,  22528, 1 };
    segs.s[4] = { W(1,1), 128, 128, 128,  43008, 0 };
    segs.s[5] = { W(1,2), 256, 128, 128,  59392, 0 };
    segs.s[6] = { W(2,0), 128, 131, 160,  92160, 1 };
    segs.s[7] = { W(2,1), 128, 128, 128, 112640, 0 };
    segs.s[8] = { W(2,2), 256, 128, 128, 129024, 0 };

    convert_weights_kernel<<<64, 256, 0, stream>>>(segs, wbf);
    transpose_feat_kernel<<<Bn * (Np / 64), 256, 0, stream>>>(features, featT);
    gather_newxyz_kernel<<<(Bn * Mq + 255) / 256, 256, 0, stream>>>(
        points, indices, nxyz, out_xyz, out_idx);
    ball_query_kernel<<<(Bn * Mq) / 8, 256, 0, stream>>>(points, nxyz, idx0, idx1, idx2);

    fused_group_mlp_kernel<16, 64, 128><<<Bn * Mq / 8, 256, 0, stream>>>(
        points, featT, nxyz, idx0, wbf, 0, 10240, 14336,
        G(0,0), Bb(0,0), G(0,1), Bb(0,1), G(0,2), Bb(0,2), out_feat, 0);
    fused_group_mlp_kernel<32, 128, 256><<<Bn * Mq / 4, 256, 0, stream>>>(
        points, featT, nxyz, idx1, wbf, 22528, 43008, 59392,
        G(1,0), Bb(1,0), G(1,1), Bb(1,1), G(1,2), Bb(1,2), out_feat, 128);
    fused_group_mlp_kernel<64, 128, 256><<<Bn * Mq / 2, 256, 0, stream>>>(
        points, featT, nxyz, idx2, wbf, 92160, 112640, 129024,
        G(2,0), Bb(2,0), G(2,1), Bb(2,1), G(2,2), Bb(2,2), out_feat, 384);
}
